// ScaledDotProductAttention_6837587935792
// MI455X (gfx1250) — matrix operand order repaired and checked by reading, not compiled, not run on hardware
//
#include <hip/hip_runtime.h>
#include <hip/hip_fp16.h>

typedef _Float16     v16h __attribute__((ext_vector_type(16)));
typedef _Float16     v8h  __attribute__((ext_vector_type(8)));
typedef float        v8f  __attribute__((ext_vector_type(8)));
typedef unsigned int v4u  __attribute__((ext_vector_type(4)));

#define NB    64      // batch
#define LQ    2048
#define LK    2048
#define DK    128
#define DV    128
#define BM    64      // q rows per block
#define BN    64      // k cols per tile
#define NTHR  256
#define QS_   136     // padded LDS strides (f16 elements)
#define KS_   136
#define VTS_  72
#define PS_   72
#define MS_   64      // mask tile stride (bytes)
#define SCALE 0.08838834764831845f   // 1/sqrt(128)
#define NEGINF (-1.0e30f)

// A-fragment (16x32 f16): lane 0-15 -> K {0..7,16..23}, lane 16-31 -> +8.
// p points at row*stride + kk + (lane>=16 ? 8 : 0); needs 16B alignment.
__device__ __forceinline__ v16h frag_a(const _Float16* p) {
  v8h lo = *(const v8h*)(p);
  v8h hv = *(const v8h*)(p + 16);
  return __builtin_shufflevector(lo, hv, 0,1,2,3,4,5,6,7,8,9,10,11,12,13,14,15);
}
// B-fragment (32x16 f16): lane 0-15 -> K {0..7,16..23}, lane 16-31 -> +8.
// p points at row*stride + kk + (lane>=16 ? 8 : 0); needs 16B alignment.
__device__ __forceinline__ v16h frag_b(const _Float16* p) {
  v8h lo = *(const v8h*)(p);
  v8h hv = *(const v8h*)(p + 16);
  return __builtin_shufflevector(lo, hv, 0,1,2,3,4,5,6,7,8,9,10,11,12,13,14,15);
}

__global__ __launch_bounds__(NTHR)
void ScaledDotProductAttention_6837587935792_kernel(
    const float* __restrict__ q, const float* __restrict__ k,
    const float* __restrict__ v, const unsigned char* __restrict__ msk,
    float* __restrict__ out, float* __restrict__ attn)
{
  __shared__ __align__(16) _Float16 Qs[BM * QS_];
  __shared__ __align__(16) _Float16 Ks[BN * KS_];
  __shared__ __align__(16) _Float16 Vt[DV * VTS_];       // V transposed: [d][kn]
  __shared__ __align__(16) _Float16 Ps[BM * PS_];        // probability re-staging
  __shared__ __align__(16) unsigned char Ms[BM * MS_];   // mask tile
  __shared__ float part_m[2][BM];
  __shared__ float part_l[2][BM];
  __shared__ float row_m[BM];
  __shared__ float row_l[BM];

  const int b    = blockIdx.y;
  const int q0   = blockIdx.x * BM;
  const int tid  = threadIdx.x;
  const int wave = tid >> 5;
  const int lane = tid & 31;
  const int nlo  = lane & 15;
  const int hiA  = (lane >> 4) << 3;   // 0 or 8  (A-frag K offset / C-row offset)
  const int hiB  = (lane >> 4) << 4;   // 0 or 16 (unused; B frags use hiA)
  const int rw   = (wave & 3) << 4;    // 0,16,32,48 : row group
  const int cg   = wave >> 2;          // 0/1 : column group

  const size_t qbase = (size_t)b * LQ * DK;
  const size_t kbase = (size_t)b * LK * DK;
  const size_t vbase = (size_t)b * LK * DV;
  const size_t mblk  = ((size_t)b * LQ + (size_t)q0) * (size_t)LK; // mask/attn tile base

  // cooperative mask-tile loader coordinates (one 16B NT load per thread)
  const int mrr = tid >> 2;          // 0..63 row in tile
  const int mcc = (tid & 3) << 4;    // 0,16,32,48

  // attn linear row offsets for the 8 rows this lane owns (C layout)
  size_t mrow[8];
  #pragma unroll
  for (int r = 0; r < 8; ++r)
    mrow[r] = mblk + (size_t)(rw + hiA + r) * (size_t)LK;

  // ---- stage Q tile once: f32 -> f16 ----
  for (int i = tid; i < BM * DK / 4; i += NTHR) {
    const int idx = i << 2;
    const int rr = idx >> 7, cc = idx & 127;
    float4 f = *(const float4*)(q + qbase + (size_t)(q0 + rr) * DK + cc);
    _Float16* d = &Qs[rr * QS_ + cc];
    d[0] = (_Float16)f.x; d[1] = (_Float16)f.y;
    d[2] = (_Float16)f.z; d[3] = (_Float16)f.w;
  }

  // =================== PASS 1: row max + sum of exp ===================
  float mrun[8], lrun[8];
  #pragma unroll
  for (int r = 0; r < 8; ++r) { mrun[r] = NEGINF; lrun[r] = 0.0f; }

  for (int n0 = 0; n0 < LK; n0 += BN) {
    __syncthreads();
    if (n0 + BN < LK)
      __builtin_prefetch(k + kbase + (size_t)(n0 + BN) * DK + (size_t)tid * 32, 0, 0);
    for (int i = tid; i < BN * DK / 4; i += NTHR) {
      const int idx = i << 2;
      const int rr = idx >> 7, cc = idx & 127;
      float4 f = *(const float4*)(k + kbase + (size_t)(n0 + rr) * DK + cc);
      _Float16* d = &Ks[rr * KS_ + cc];
      d[0] = (_Float16)f.x; d[1] = (_Float16)f.y;
      d[2] = (_Float16)f.z; d[3] = (_Float16)f.w;
    }
    // mask tile: coalesced 16B non-temporal load per thread (streamed data)
    {
      const v4u mv = __builtin_nontemporal_load(
          (const v4u*)(msk + mblk + (size_t)mrr * LK + (size_t)(n0 + mcc)));
      *(v4u*)&Ms[mrr * MS_ + mcc] = mv;
    }
    __syncthreads();

    #pragma unroll
    for (int c = 0; c < 2; ++c) {
      const int cbase = cg * 32 + c * 16;
      v8f acc = {};
      #pragma unroll
      for (int kk = 0; kk < DK; kk += 32) {
        v16h a  = frag_a(&Qs[(rw    + nlo) * QS_ + kk + hiA]);
        v16h bm = frag_b(&Ks[(cbase + nlo) * KS_ + kk + hiA]);
        acc = __builtin_amdgcn_wmma_f32_16x16x32_f16(false, a, false, bm,
                                                     (short)0, acc, false, false);
      }
      #pragma unroll
      for (int r = 0; r < 8; ++r) {
        float s = acc[r] * SCALE;
        if (!Ms[(rw + hiA + r) * MS_ + cbase + nlo]) s = NEGINF;
        const float mo = mrun[r];
        const float mn = fmaxf(mo, s);
        lrun[r] = lrun[r] * __expf(mo - mn) + __expf(s - mn);
        mrun[r] = mn;
      }
    }
  }

  // lane reduction within each 16-lane half (same rows, different cols)
  #pragma unroll
  for (int r = 0; r < 8; ++r) {
    float m = mrun[r], l = lrun[r];
    #pragma unroll
    for (int off = 1; off < 16; off <<= 1) {
      const float om = __shfl_xor(m, off, 32);
      const float ol = __shfl_xor(l, off, 32);
      const float mn = fmaxf(m, om);
      l = l * __expf(m - mn) + ol * __expf(om - mn);
      m = mn;
    }
    mrun[r] = m; lrun[r] = l;
  }
  if ((lane & 15) == 0) {
    #pragma unroll
    for (int r = 0; r < 8; ++r) {
      part_m[cg][rw + hiA + r] = mrun[r];
      part_l[cg][rw + hiA + r] = lrun[r];
    }
  }
  __syncthreads();
  if (tid < BM) {
    const float m0 = part_m[0][tid], m1 = part_m[1][tid];
    const float l0 = part_l[0][tid], l1 = part_l[1][tid];
    const float M = fmaxf(m0, m1);
    const float L = l0 * __expf(m0 - M) + l1 * __expf(m1 - M);
    row_m[tid] = M;
    row_l[tid] = L;
  }
  __syncthreads();

  float rm[8], rinv[8];
  #pragma unroll
  for (int r = 0; r < 8; ++r) {
    rm[r]   = row_m[rw + hiA + r];
    rinv[r] = 1.0f / row_l[rw + hiA + r];
  }

  // =================== PASS 2: write attn, accumulate O = P·V ===================
  v8f accO[4] = {};
  for (int n0 = 0; n0 < LK; n0 += BN) {
    __syncthreads();
    if (n0 + BN < LK) {
      __builtin_prefetch(k + kbase + (size_t)(n0 + BN) * DK + (size_t)tid * 32, 0, 0);
      __builtin_prefetch(v + vbase + (size_t)(n0 + BN) * DV + (size_t)tid * 32, 0, 0);
    }
    for (int i = tid; i < BN * DK / 4; i += NTHR) {
      const int idx = i << 2;
      const int rr = idx >> 7, cc = idx & 127;
      float4 f = *(const float4*)(k + kbase + (size_t)(n0 + rr) * DK + cc);
      _Float16* d = &Ks[rr * KS_ + cc];
      d[0] = (_Float16)f.x; d[1] = (_Float16)f.y;
      d[2] = (_Float16)f.z; d[3] = (_Float16)f.w;
    }
    for (int i = tid; i < BN * DV / 4; i += NTHR) {
      const int idx = i << 2;
      const int rr = idx >> 7, cc = idx & 127;
      float4 f = *(const float4*)(v + vbase + (size_t)(n0 + rr) * DV + cc);
      Vt[(cc + 0) * VTS_ + rr] = (_Float16)f.x;
      Vt[(cc + 1) * VTS_ + rr] = (_Float16)f.y;
      Vt[(cc + 2) * VTS_ + rr] = (_Float16)f.z;
      Vt[(cc + 3) * VTS_ + rr] = (_Float16)f.w;
    }
    {
      const v4u mv = __builtin_nontemporal_load(
          (const v4u*)(msk + mblk + (size_t)mrr * LK + (size_t)(n0 + mcc)));
      *(v4u*)&Ms[mrr * MS_ + mcc] = mv;
    }
    __syncthreads();

    #pragma unroll
    for (int c = 0; c < 2; ++c) {
      const int cbase = cg * 32 + c * 16;
      v8f acc = {};
      #pragma unroll
      for (int kk = 0; kk < DK; kk += 32) {
        v16h a  = frag_a(&Qs[(rw    + nlo) * QS_ + kk + hiA]);
        v16h bm = frag_b(&Ks[(cbase + nlo) * KS_ + kk + hiA]);
        acc = __builtin_amdgcn_wmma_f32_16x16x32_f16(false, a, false, bm,
                                                     (short)0, acc, false, false);
      }
      #pragma unroll
      for (int r = 0; r < 8; ++r) {
        float s = acc[r] * SCALE;
        if (!Ms[(rw + hiA + r) * MS_ + cbase + nlo]) s = NEGINF;
        const float p = __expf(s - rm[r]) * rinv[r];
        // dominant store traffic: stream past L2 (non-temporal)
        __builtin_nontemporal_store(p, &attn[mrow[r] + (size_t)(n0 + cbase + nlo)]);
        Ps[(rw + hiA + r) * PS_ + cbase + nlo] = (_Float16)p;
      }
    }
    __syncthreads();

    #pragma unroll
    for (int kc = 0; kc < 2; ++kc) {
      v16h a = frag_a(&Ps[(rw + nlo) * PS_ + kc * 32 + hiA]);
      #pragma unroll
      for (int j = 0; j < 4; ++j) {
        v16h bm = frag_b(&Vt[(cg * 64 + j * 16 + nlo) * VTS_ + kc * 32 + hiA]);
        accO[j] = __builtin_amdgcn_wmma_f32_16x16x32_f16(false, a, false, bm,
                                                         (short)0, accO[j], false, false);
      }
    }
  }

  // ---- epilogue: O (already normalized, since p was normalized) ----
  #pragma unroll
  for (int j = 0; j < 4; ++j) {
    #pragma unroll
    for (int r = 0; r < 8; ++r) {
      __builtin_nontemporal_store(
          accO[j][r],
          &out[((size_t)b * LQ + (size_t)(q0 + rw + hiA + r)) * DV + cg * 64 + j * 16 + nlo]);
    }
  }
}

extern "C" void kernel_launch(void* const* d_in, const int* in_sizes, int n_in,
                              void* d_out, int out_size, void* d_ws, size_t ws_size,
                              hipStream_t stream) {
  (void)in_sizes; (void)n_in; (void)out_size; (void)d_ws; (void)ws_size;
  const float* q = (const float*)d_in[0];
  const float* k = (const float*)d_in[1];
  const float* v = (const float*)d_in[2];
  const unsigned char* msk = (const unsigned char*)d_in[3];

  float* out  = (float*)d_out;                              // [B, Lq, Dv]
  float* attn = out + (size_t)NB * LQ * DV;                 // [B, Lq, Lk]

  dim3 grid(LQ / BM, NB);
  ScaledDotProductAttention_6837587935792_kernel<<<grid, NTHR, 0, stream>>>(
      q, k, v, msk, out, attn);
}
